// RestrictedNN_18915035972223
// MI455X (gfx1250) — matrix operand order repaired and checked by reading, not compiled, not run on hardware
//
#include <hip/hip_runtime.h>

// ---------------------------------------------------------------------------
// Fused RestrictedNN forward for MI455X (gfx1250, wave32, WMMA bf16).
//   leaf (folded into weights) -> mid (16 x [B,256]x[256,128] + tanh)
//   -> root ([B,2048]x[2048,256])
// Roofline: ~276 MB compulsory HBM -> ~12 us @ 23.3 TB/s; needs ~3 PFLOPS
// -> v_wmma_f32_16x16x32_bf16 (f32 accumulate). Wave-pair split keeps VGPRs
// ~150/wave for 3+ waves/SIMD; next-g weight prefetch (global_prefetch_b8)
// warms WGP$ a full iteration ahead; ping-pong LDS -> 1 barrier per g.
// ---------------------------------------------------------------------------

typedef __attribute__((ext_vector_type(16))) __bf16 v16bf;
typedef __attribute__((ext_vector_type(8)))  __bf16 v8bf;
typedef __attribute__((ext_vector_type(8)))  float  v8f;
typedef __attribute__((ext_vector_type(4)))  float  v4f;

#define IN_DIM 4096
#define N_MID  16
#define OUT_N  256
#define BATCH  16384
#define BT     64      // batch rows per block (4 pairs x 16 rows, 8 waves)

// Workspace layout (bytes):
//   [0, 1MB)      Wm  : bf16 B-fragments of folded mid weights
//                  frag id = g*64 + ks*8 + nt  (g<16, ks<8, nt<8), 512 bf16/frag
//   [1MB, 2MB)    Wr  : bf16 B-fragments of root weights
//                  frag id = kb*16 + nt        (kb<64, nt<16),     512 bf16/frag
//   [2MB, 2MB+8K) beff: f32[16*128] effective mid bias (pbias folded in)
// Fragment element order: lane l, elem p -> K = (l/16)*8 + (p%8) + (p/8)*16, N = ntile*16 + l%16

// ---- prep: fold leaf diag weights into mid weights, swizzle to B-frag layout
__global__ void prep_wm(const float* __restrict__ w_inp,
                        const float* __restrict__ W_mid,
                        __bf16* __restrict__ Wm) {
  int bid  = blockIdx.x;            // g*64 + ks*8 + nt
  int g    = bid >> 6;
  int ks   = (bid >> 3) & 7;
  int nt   = bid & 7;
  int lane = threadIdx.x;           // 0..31
  int lrow = lane & 15, hs = lane >> 4;
  int n    = nt * 16 + lrow;
  __bf16* dst = Wm + ((size_t)bid * 32 + lane) * 16;
#pragma unroll
  for (int p = 0; p < 16; ++p) {
    int k = ks * 32 + hs * 8 + (p & 7) + ((p >> 3) << 4); // 0..255 within module
    float w = w_inp[(g * 4 + (k >> 6)) * 64 + (k & 63)]; // leaf diagonal entry
    float v = w * W_mid[((size_t)(g * 256 + k)) * 128 + n];
    dst[p] = (__bf16)v;
  }
}

// ---- prep: swizzle root weights to B-frag layout
__global__ void prep_wr(const float* __restrict__ W_root,
                        __bf16* __restrict__ Wr) {
  int bid  = blockIdx.x;            // kb*16 + nt
  int kb   = bid >> 4;
  int nt   = bid & 15;
  int lane = threadIdx.x;
  int lrow = lane & 15, hs = lane >> 4;
  int n    = nt * 16 + lrow;
  __bf16* dst = Wr + ((size_t)bid * 32 + lane) * 16;
#pragma unroll
  for (int p = 0; p < 16; ++p) {
    int k = kb * 32 + hs * 8 + (p & 7) + ((p >> 3) << 4);  // 0..2047
    dst[p] = (__bf16)W_root[(size_t)k * 256 + n];
  }
}

// ---- prep: effective mid bias  b_eff = b_mid + sum_i pbias(leaf(i)) * W_mid
__global__ void prep_beff(const float* __restrict__ pbias,
                          const float* __restrict__ W_mid,
                          const float* __restrict__ b_mid,
                          float* __restrict__ beff) {
  int t = blockIdx.x * blockDim.x + threadIdx.x;   // 0..2047
  int g = t >> 7, o = t & 127;
  float acc = b_mid[t];
  for (int i = 0; i < 256; ++i)
    acc += pbias[g * 4 + (i >> 6)] * W_mid[((size_t)(g * 256 + i)) * 128 + o];
  beff[t] = acc;
}

__device__ __forceinline__ float tanh_fast(float x) {
  float e = __expf(2.0f * x);      // v_exp_f32 path
  return (e - 1.0f) / (e + 1.0f);
}

// ---- fused mid+root kernel.
// A wave PAIR owns 16 batch rows; sub-wave s does mid N-tiles [4s,4s+4) and
// root N-tiles [8s, 8s+8). The pair shares a ping-ponged 16x128 bf16 LDS tile.
__global__ __launch_bounds__(256)
void fused_nn(const float* __restrict__ x,
              const __bf16* __restrict__ Wm,
              const __bf16* __restrict__ Wr,
              const float* __restrict__ beff,
              const float* __restrict__ b_root,
              float* __restrict__ out) {
  __shared__ __bf16 lds_mid[2][4][16 * 128];  // ping-pong per-pair staging (32 KB)

  int lane = threadIdx.x & 31;
  int wave = threadIdx.x >> 5;     // 0..7
  int pair = wave >> 1;            // 0..3
  int sub  = wave & 1;             // 0/1: N-split within the pair
  int lrow = lane & 15;
  int hs   = lane >> 4;
  int r0   = blockIdx.x * BT + pair * 16;
  const float* xrow = x + (size_t)(r0 + lrow) * IN_DIM;

  v8f acc_out[8];                  // root C-tiles: 16 x 128 f32 (this sub's half)
#pragma unroll
  for (int i = 0; i < 8; ++i)
#pragma unroll
    for (int j = 0; j < 8; ++j) acc_out[i][j] = 0.0f;

  for (int g = 0; g < N_MID; ++g) {
    // -------- prefetch next g's weight fragment slices into WGP$ -----------
    // Each g-slice of Wm and Wr is 64 KB contiguous; 256 threads x 2 lines
    // (128 B apart) cover each slice. Wrap at 16 to stay in-bounds.
    {
      int ng = (g + 1) & 15;
      const char* pm = (const char*)(Wm + (size_t)ng * 64 * 512) + threadIdx.x * 128;
      const char* pr = (const char*)(Wr + (size_t)ng * 64 * 512) + threadIdx.x * 128;
      __builtin_prefetch(pm,         0, 3);
      __builtin_prefetch(pm + 32768, 0, 3);
      __builtin_prefetch(pr,         0, 3);
      __builtin_prefetch(pr + 32768, 0, 3);
    }

    // -------- mid GEMM, this sub's half: [16x256] x [256x64] ---------------
    v8f accm[4];
#pragma unroll
    for (int i = 0; i < 4; ++i)
#pragma unroll
      for (int j = 0; j < 8; ++j) accm[i][j] = 0.0f;

#pragma unroll
    for (int ks = 0; ks < 8; ++ks) {
      // A fragment: 16x32 bf16 from x (f32->bf16 on the fly).
      // Lane layout: lanes 0-15 p0-7=K0-7, p8-15=K16-23; lanes 16-31 +8.
      int kbase = g * 256 + ks * 32;
      const float* p0 = xrow + kbase + hs * 8;
      const float* p1 = xrow + kbase + 16 + hs * 8;
      v4f f0 = *(const v4f*)p0;
      v4f f1 = *(const v4f*)(p0 + 4);
      v4f f2 = *(const v4f*)p1;
      v4f f3 = *(const v4f*)(p1 + 4);
      v16bf a;
#pragma unroll
      for (int i = 0; i < 4; ++i) {
        a[i]      = (__bf16)f0[i];
        a[4 + i]  = (__bf16)f1[i];
        a[8 + i]  = (__bf16)f2[i];
        a[12 + i] = (__bf16)f3[i];
      }
      const __bf16* wmBase =
          Wm + (((size_t)(g * 64 + ks * 8 + sub * 4)) * 32 + lane) * 16;
#pragma unroll
      for (int j = 0; j < 4; ++j) {
        v16bf b = *(const v16bf*)(wmBase + (size_t)j * 512);
        accm[j] = __builtin_amdgcn_wmma_f32_16x16x32_bf16(
            false, a, false, b, (short)0, accm[j], false, false);
      }
    }

    // -------- bias + tanh, stage this sub's 16x64 half into shared tile ----
    __bf16* myLds = lds_mid[g & 1][pair];
#pragma unroll
    for (int j = 0; j < 4; ++j) {
      int ntg = sub * 4 + j;                 // global mid N-tile 0..7
      float bias = beff[g * 128 + ntg * 16 + lrow];
#pragma unroll
      for (int e = 0; e < 8; ++e) {
        int m = e + hs * 8;                  // C-layout: VGPR e, lane half -> M
        float v = tanh_fast(accm[j][e] + bias);
        myLds[m * 128 + ntg * 16 + lrow] = (__bf16)v;
      }
    }
    __syncthreads();   // writes(g, buf g&1) visible; next writes use other buf

    // -------- root K-block g: out += mid_g [16x128] x Wr_g [128x(this half)]
#pragma unroll
    for (int ks2 = 0; ks2 < 4; ++ks2) {
      const __bf16* l0 = myLds + lrow * 128 + ks2 * 32 + hs * 8;
      v8bf c0 = *(const v8bf*)l0;
      v8bf c1 = *(const v8bf*)(l0 + 16);
      v16bf a2;
#pragma unroll
      for (int i = 0; i < 8; ++i) { a2[i] = c0[i]; a2[8 + i] = c1[i]; }
      int kb = g * 4 + ks2;                  // root K tile index (0..63)
      const __bf16* wrBase =
          Wr + (((size_t)(kb * 16 + sub * 8)) * 32 + lane) * 16;
#pragma unroll
      for (int j = 0; j < 8; ++j) {
        v16bf b = *(const v16bf*)(wrBase + (size_t)j * 512);
        acc_out[j] = __builtin_amdgcn_wmma_f32_16x16x32_bf16(
            false, a2, false, b, (short)0, acc_out[j], false, false);
      }
    }
    // no second barrier: ping-pong buffer isolates next g's writes
  }

  // -------- epilogue: + b_root, store f32 (this sub's 128 output cols) -----
#pragma unroll
  for (int j = 0; j < 8; ++j) {
    int n = (sub * 8 + j) * 16 + lrow;
    float bias = b_root[n];
#pragma unroll
    for (int e = 0; e < 8; ++e) {
      int row = r0 + e + hs * 8;
      out[(size_t)row * OUT_N + n] = acc_out[j][e] + bias;
    }
  }
}

extern "C" void kernel_launch(void* const* d_in, const int* in_sizes, int n_in,
                              void* d_out, int out_size, void* d_ws, size_t ws_size,
                              hipStream_t stream) {
  (void)in_sizes; (void)n_in; (void)out_size; (void)ws_size;
  const float* x      = (const float*)d_in[0];
  const float* w_inp  = (const float*)d_in[1];
  const float* pbias  = (const float*)d_in[2];
  const float* W_mid  = (const float*)d_in[3];
  const float* b_mid  = (const float*)d_in[4];
  const float* W_root = (const float*)d_in[5];
  const float* b_root = (const float*)d_in[6];
  float* out = (float*)d_out;

  __bf16* Wm   = reinterpret_cast<__bf16*>(d_ws);
  __bf16* Wr   = reinterpret_cast<__bf16*>((char*)d_ws + (1u << 20));
  float*  beff = reinterpret_cast<float*>((char*)d_ws + (2u << 20));

  prep_wm  <<<1024, 32, 0, stream>>>(w_inp, W_mid, Wm);
  prep_wr  <<<1024, 32, 0, stream>>>(W_root, Wr);
  prep_beff<<<8,   256, 0, stream>>>(pbias, W_mid, b_mid, beff);
  fused_nn <<<BATCH / BT, 256, 0, stream>>>(x, Wm, Wr, beff, b_root, out);
}
